// MultiHeadLatentAttention_71622874628312
// MI455X (gfx1250) — matrix operand order repaired and checked by reading, not compiled, not run on hardware
//
#include <hip/hip_runtime.h>

// ---------------------------------------------------------------------------
// MLA forward for MI455X (gfx1250): all GEMMs + attention via
// v_wmma_f32_16x16x32_bf16, flash-style attention (no SxS materialization).
// Round 2: flash kernel de-spilled (Q in LDS, score-transposed softmax with
// 2 shuffles/chunk instead of 64), GEMM software-pipelined.
// ---------------------------------------------------------------------------

typedef __bf16 bf16_t;
typedef __attribute__((ext_vector_type(16))) __bf16 v16bf;
typedef __attribute__((ext_vector_type(8)))  __bf16 v8bf;
typedef __attribute__((ext_vector_type(8)))  float  v8f;

static constexpr int Bc   = 2;
static constexpr int Sc   = 2048;
static constexpr int Dc   = 2048;
static constexpr int Hc   = 16;
static constexpr int DNc  = 64;
static constexpr int DRc  = 64;
static constexpr int DQKc = 128;   // DN + DR
static constexpr int DVc  = 128;
static constexpr int QLRc = 1536;
static constexpr int KVLRc= 1024;
static constexpr int Mrows = Bc * Sc;          // 4096
static constexpr int HDc   = Hc * DVc;         // 2048

__device__ __forceinline__ v8f wmma_bf(v16bf a, v16bf b, v8f c) {
  // (neg_a, A, neg_b, B, c_mod, C, reuse_a, reuse_b)
  return __builtin_amdgcn_wmma_f32_16x16x32_bf16(false, a, false, b, (short)0, c, false, false);
}

// ------------------------------ f32 -> bf16 --------------------------------
__global__ void k_cast_bf16(const float* __restrict__ in, bf16_t* __restrict__ out, long n) {
  long i = (long)blockIdx.x * blockDim.x + threadIdx.x;
  long stride = (long)gridDim.x * blockDim.x;
  for (; i < n; i += stride) out[i] = (bf16_t)in[i];
}

// ------------------------------ GEMM: C = A * W^T --------------------------
// A: (M,K) bf16 row-major, W: (N,K) bf16 row-major. One wave per 16x64 tile.
// A-frag: lane holds row (lane&15), 16 contiguous K at hi*16.
// B-frag: lane holds W row (n0+j*16+(lane&15)), 16 contiguous K at hi*16.
// One-stage software pipeline: fragments for k+32 are loaded before the k
// WMMAs consume the current ones.
template <bool BF16OUT>
__global__ void k_gemm(const bf16_t* __restrict__ A, const bf16_t* __restrict__ W,
                       float* __restrict__ Cf, bf16_t* __restrict__ Cb,
                       int N, int K) {
  const int lane = threadIdx.x;
  const int ln   = lane & 15;
  const int hi   = lane >> 4;
  const int m0   = blockIdx.y << 4;
  const int n0   = blockIdx.x << 6;

  const bf16_t* Ab = A + (size_t)(m0 + ln) * K;

  auto ldA = [&](int k0) {
    union { v16bf v; v8bf h[2]; } au;
    au.h[0] = *(const v8bf*)(Ab + k0 + hi * 16);
    au.h[1] = *(const v8bf*)(Ab + k0 + hi * 16 + 8);
    return au.v;
  };
  auto ldB = [&](int k0, int j) {
    return *(const v16bf*)(W + (size_t)(n0 + j * 16 + ln) * K + k0 + hi * 16);
  };

  v8f acc[4];
  #pragma unroll
  for (int j = 0; j < 4; ++j) acc[j] = (v8f)0.0f;

  v16bf a0 = ldA(0);
  v16bf b0[4];
  #pragma unroll
  for (int j = 0; j < 4; ++j) b0[j] = ldB(0, j);

  for (int k0 = 0; k0 < K; k0 += 32) {
    const int kn = (k0 + 32 < K) ? k0 + 32 : k0;   // last iter: harmless reload
    __builtin_prefetch(Ab + kn + 256, 0, 0);       // global_prefetch_b8
    v16bf a1 = ldA(kn);
    v16bf b1[4];
    #pragma unroll
    for (int j = 0; j < 4; ++j) b1[j] = ldB(kn, j);
    #pragma unroll
    for (int j = 0; j < 4; ++j) acc[j] = wmma_bf(a0, b0[j], acc[j]);
    a0 = a1;
    #pragma unroll
    for (int j = 0; j < 4; ++j) b0[j] = b1[j];
  }

  #pragma unroll
  for (int j = 0; j < 4; ++j) {
    #pragma unroll
    for (int r = 0; r < 8; ++r) {
      size_t idx = (size_t)(m0 + r + hi * 8) * N + n0 + j * 16 + ln;
      if constexpr (BF16OUT) Cb[idx] = (bf16_t)acc[j][r];
      else                   Cf[idx] = acc[j][r];
    }
  }
}

// ------------------------------ RMSNorm (f32 in, bf16 out) -----------------
__global__ void k_rmsnorm(const float* __restrict__ in, const float* __restrict__ w,
                          bf16_t* __restrict__ out, int L, int ldin) {
  const int row = blockIdx.x;
  const float* x = in + (size_t)row * ldin;
  __shared__ float red[256];
  float s = 0.f;
  for (int i = threadIdx.x; i < L; i += 256) { float v = x[i]; s += v * v; }
  red[threadIdx.x] = s;
  __syncthreads();
  for (int o = 128; o > 0; o >>= 1) {
    if (threadIdx.x < o) red[threadIdx.x] += red[threadIdx.x + o];
    __syncthreads();
  }
  const float r = rsqrtf(red[0] / (float)L + 1e-6f);
  bf16_t* o = out + (size_t)row * L;
  for (int i = threadIdx.x; i < L; i += 256) o[i] = (bf16_t)(x[i] * r * w[i]);
}

// ------------------------------ k_pe RoPE (from kv f32 tail) ---------------
__global__ void k_kpe_rope(const float* __restrict__ kv, const float* __restrict__ fc,
                           const float* __restrict__ fs, bf16_t* __restrict__ kpe) {
  const int row = blockIdx.x;            // b*S + s
  const int s   = row % Sc;
  const int i   = threadIdx.x;           // 0..31 (pairs)
  const float* src = kv + (size_t)row * (KVLRc + DRc) + KVLRc;
  const float x0 = src[2 * i], x1 = src[2 * i + 1];
  const float c = fc[s * 32 + i], sn = fs[s * 32 + i];
  kpe[(size_t)row * DRc + 2 * i]     = (bf16_t)(x0 * c - x1 * sn);
  kpe[(size_t)row * DRc + 2 * i + 1] = (bf16_t)(x0 * sn + x1 * c);
}

// ------------------------------ pack Q: rope + scale, (B,H,S,DQK) ----------
__global__ void k_pack_q(const float* __restrict__ q, const float* __restrict__ fc,
                         const float* __restrict__ fs, bf16_t* __restrict__ Qp) {
  const int idx = blockIdx.x;            // row*H + h
  const int h   = idx % Hc;
  const int row = idx / Hc;              // b*S + s
  const int b   = row / Sc, s = row % Sc;
  const float* src = q + ((size_t)row * Hc + h) * DQKc;
  bf16_t* dst = Qp + (((size_t)(b * Hc + h) * Sc) + s) * DQKc;
  const float scale = 0.08838834764831845f;   // 1/sqrt(DQK=128)
  const int t = threadIdx.x;             // 64 threads
  dst[t] = (bf16_t)(src[t] * scale);
  if (t < 32) {
    const float x0 = src[DNc + 2 * t], x1 = src[DNc + 2 * t + 1];
    const float c = fc[s * 32 + t], sn = fs[s * 32 + t];
    dst[DNc + 2 * t]     = (bf16_t)((x0 * c - x1 * sn) * scale);
    dst[DNc + 2 * t + 1] = (bf16_t)((x0 * sn + x1 * c) * scale);
  }
}

// ------------------------------ pack K (B,H,S,DQK) and V^T (B,H,DV,S) ------
__global__ void k_pack_kv(const bf16_t* __restrict__ kv2, const bf16_t* __restrict__ kpe,
                          bf16_t* __restrict__ Kp, bf16_t* __restrict__ Vt) {
  const int idx = blockIdx.x;            // row*H + h
  const int h   = idx % Hc;
  const int row = idx / Hc;
  const int b   = row / Sc, s = row % Sc;
  const bf16_t* src = kv2 + ((size_t)row * Hc + h) * (DNc + DVc);
  const int t = threadIdx.x;             // 128 threads
  const size_t kbase = (((size_t)(b * Hc + h) * Sc) + s) * DQKc;
  Kp[kbase + t] = (t < DNc) ? src[t] : kpe[(size_t)row * DRc + (t - DNc)];
  Vt[(((size_t)(b * Hc + h) * DVc) + t) * Sc + s] = src[DNc + t];
}

// ------------------------------ flash attention ----------------------------
// One wave per 16-query tile per (b,h). Q lives in LDS (no long-lived frags).
// Per 32-key chunk:
//   scores (C-layout) -> LDS as raw f32 -> each lane re-reads its A-frag
//   slice (row lane&15), computes mask+max+exp+sum locally, ONE shfl_xor(16)
//   per reduction, converts P directly into A-frag registers, then 8 WMMAs
//   accumulate O(16x128). O rescale factors redistributed with 8 shuffles.
__global__ void k_flash(const bf16_t* __restrict__ Qp, const bf16_t* __restrict__ Kp,
                        const bf16_t* __restrict__ Vt, bf16_t* __restrict__ Out) {
  const int bh   = blockIdx.y;           // b*H + h
  const int m0   = blockIdx.x << 4;
  const int lane = threadIdx.x;
  const int ln   = lane & 15;
  const int hi   = lane >> 4;
  const int b    = bh / Hc, h = bh % Hc;

  const bf16_t* Q = Qp + (size_t)bh * Sc * DQKc;
  const bf16_t* K = Kp + (size_t)bh * Sc * DQKc;
  const bf16_t* V = Vt + (size_t)bh * DVc * Sc;

  __shared__ __align__(64) bf16_t Qs[16][DQKc];   // 4 KB
  __shared__ __align__(64) float  Sl[16][32];     // 2 KB

  // stage Q tile into LDS (each lane copies half a row)
  {
    const bf16_t* qrow = Q + (size_t)(m0 + ln) * DQKc + hi * 64;
    #pragma unroll
    for (int c = 0; c < 8; ++c)
      *(v8bf*)&Qs[ln][hi * 64 + c * 8] = *(const v8bf*)(qrow + c * 8);
  }
  __syncthreads();

  v8f O[8];
  #pragma unroll
  for (int j = 0; j < 8; ++j) O[j] = (v8f)0.0f;
  float mr = -1e30f;   // running max for row (m0 + ln), replicated in both halves
  float lr = 0.f;      // running sum for row (m0 + ln)

  for (int t0 = 0; t0 <= m0 + 15; t0 += 32) {
    // ---- scores: Q(16x128) @ K^T for keys [t0, t0+32)
    v8f sc0 = (v8f)0.0f, sc1 = (v8f)0.0f;
    #pragma unroll
    for (int kc = 0; kc < 4; ++kc) {
      union { v16bf v; v8bf hh[2]; } au;
      au.hh[0] = *(const v8bf*)&Qs[ln][kc * 32 + hi * 16];
      au.hh[1] = *(const v8bf*)&Qs[ln][kc * 32 + hi * 16 + 8];
      v16bf k0v = *(const v16bf*)(K + (size_t)(t0 + ln) * DQKc + kc * 32 + hi * 16);
      sc0 = wmma_bf(au.v, k0v, sc0);
      v16bf k1v = *(const v16bf*)(K + (size_t)(t0 + 16 + ln) * DQKc + kc * 32 + hi * 16);
      sc1 = wmma_bf(au.v, k1v, sc1);
    }

    // ---- C-layout -> LDS (raw f32 scores)
    __syncthreads();   // WAR vs previous iteration's reads
    #pragma unroll
    for (int r = 0; r < 8; ++r) {
      Sl[r + hi * 8][ln]      = sc0[r];
      Sl[r + hi * 8][16 + ln] = sc1[r];
    }
    __syncthreads();

    // ---- row phase: lane owns row (m0+ln), its A-frag slice of 16 keys
    v8f s0 = *(const v8f*)&Sl[ln][hi * 16];       // keys t0 + hi*16 + i
    v8f s1 = *(const v8f*)&Sl[ln][hi * 16 + 8];   // keys t0 + hi*16 + 8 + i
    const int srow = m0 + ln;
    if (t0 + 31 > m0) {
      #pragma unroll
      for (int i = 0; i < 8; ++i) {
        if (t0 + hi * 16 + i > srow)     s0[i] = -1e30f;
        if (t0 + hi * 16 + 8 + i > srow) s1[i] = -1e30f;
      }
    }
    float mx = fmaxf(s0[0], s1[0]);
    #pragma unroll
    for (int i = 1; i < 8; ++i) mx = fmaxf(mx, fmaxf(s0[i], s1[i]));
    mx = fmaxf(mx, __shfl_xor(mx, 16, 32));       // combine the two halves
    const float mnew  = fmaxf(mr, mx);
    const float alpha = __expf(mr - mnew);
    mr = mnew;

    v16bf pf;                                      // P in A-frag layout
    float ps = 0.f;
    #pragma unroll
    for (int i = 0; i < 8; ++i) {
      const float e0 = __expf(s0[i] - mnew);
      const float e1 = __expf(s1[i] - mnew);
      ps += e0 + e1;
      pf[i]     = (bf16_t)e0;
      pf[8 + i] = (bf16_t)e1;
    }
    ps += __shfl_xor(ps, 16, 32);
    lr = lr * alpha + ps;

    // ---- rescale O (alpha lives per-row in lanes 0..15; redistribute)
    #pragma unroll
    for (int r = 0; r < 8; ++r) {
      const float ac = __shfl(alpha, r + hi * 8, 32);
      #pragma unroll
      for (int j = 0; j < 8; ++j) O[j][r] *= ac;
    }

    // ---- O += P @ V  (V^T rows contiguous in keys)
    #pragma unroll
    for (int j = 0; j < 8; ++j) {
      v16bf vf = *(const v16bf*)(V + (size_t)(j * 16 + ln) * Sc + t0 + hi * 16);
      O[j] = wmma_bf(pf, vf, O[j]);
    }
  }

  // ---- epilogue: normalize and store (B,S,H*DV) bf16
  #pragma unroll
  for (int r = 0; r < 8; ++r) {
    const float li   = __shfl(lr, r + hi * 8, 32);
    const float linv = 1.f / li;
    const int   srow = m0 + r + hi * 8;
    #pragma unroll
    for (int j = 0; j < 8; ++j)
      Out[((size_t)(b * Sc + srow)) * HDc + h * DVc + j * 16 + ln] = (bf16_t)(O[j][r] * linv);
  }
}

// ===========================================================================
extern "C" void kernel_launch(void* const* d_in, const int* in_sizes, int n_in,
                              void* d_out, int out_size, void* d_ws, size_t ws_size,
                              hipStream_t stream) {
  const float* x      = (const float*)d_in[0];
  const float* wq_a   = (const float*)d_in[1];
  const float* qnw    = (const float*)d_in[2];
  const float* wq_b   = (const float*)d_in[3];
  const float* wkv_a  = (const float*)d_in[4];
  const float* kvnw   = (const float*)d_in[5];
  const float* wkv_b  = (const float*)d_in[6];
  const float* wo     = (const float*)d_in[7];
  // d_in[8] mask (causal, handled in-kernel), d_in[9/10] freqs, d_in[11] start_pos (0)
  const float* fcos   = (const float*)d_in[9];
  const float* fsin   = (const float*)d_in[10];
  float* out = (float*)d_out;

  char* ws = (char*)d_ws;
  size_t off = 0;
  auto alloc = [&](size_t bytes) { size_t o = off; off = (off + bytes + 255) & ~(size_t)255; return o; };

  bf16_t* x_bf    = (bf16_t*)(ws + alloc((size_t)Mrows * Dc * 2));
  bf16_t* wqa_bf  = (bf16_t*)(ws + alloc((size_t)QLRc * Dc * 2));
  bf16_t* wqb_bf  = (bf16_t*)(ws + alloc((size_t)Hc * DQKc * QLRc * 2));
  bf16_t* wkva_bf = (bf16_t*)(ws + alloc((size_t)(KVLRc + DRc) * Dc * 2));
  bf16_t* wkvb_bf = (bf16_t*)(ws + alloc((size_t)Hc * (DNc + DVc) * KVLRc * 2));
  bf16_t* wo_bf   = (bf16_t*)(ws + alloc((size_t)Dc * HDc * 2));
  float*  qa_f32  = (float*)(ws + alloc((size_t)Mrows * QLRc * 4));
  bf16_t* qa_bf   = (bf16_t*)(ws + alloc((size_t)Mrows * QLRc * 2));
  float*  q_f32   = (float*)(ws + alloc((size_t)Mrows * Hc * DQKc * 4));
  float*  kv_f32  = (float*)(ws + alloc((size_t)Mrows * (KVLRc + DRc) * 4));
  bf16_t* ckv_bf  = (bf16_t*)(ws + alloc((size_t)Mrows * KVLRc * 2));
  bf16_t* kpe_bf  = (bf16_t*)(ws + alloc((size_t)Mrows * DRc * 2));
  bf16_t* kv2_bf  = (bf16_t*)(ws + alloc((size_t)Mrows * Hc * (DNc + DVc) * 2));
  bf16_t* Qp      = (bf16_t*)(ws + alloc((size_t)Bc * Hc * Sc * DQKc * 2));
  bf16_t* Kp      = (bf16_t*)(ws + alloc((size_t)Bc * Hc * Sc * DQKc * 2));
  bf16_t* Vt      = (bf16_t*)(ws + alloc((size_t)Bc * Hc * DVc * Sc * 2));
  bf16_t* attn_bf = (bf16_t*)(ws + alloc((size_t)Mrows * HDc * 2));
  (void)ws_size; (void)in_sizes; (void)n_in; (void)out_size;

  // ---- cast inputs / weights to bf16
  k_cast_bf16<<<4096, 256, 0, stream>>>(x,     x_bf,    (long)Mrows * Dc);
  k_cast_bf16<<<2048, 256, 0, stream>>>(wq_a,  wqa_bf,  (long)QLRc * Dc);
  k_cast_bf16<<<2048, 256, 0, stream>>>(wq_b,  wqb_bf,  (long)Hc * DQKc * QLRc);
  k_cast_bf16<<<2048, 256, 0, stream>>>(wkv_a, wkva_bf, (long)(KVLRc + DRc) * Dc);
  k_cast_bf16<<<2048, 256, 0, stream>>>(wkv_b, wkvb_bf, (long)Hc * (DNc + DVc) * KVLRc);
  k_cast_bf16<<<2048, 256, 0, stream>>>(wo,    wo_bf,   (long)Dc * HDc);

  const dim3 blk32(32);

  // ---- q_a = x @ wq_a^T  (4096 x 1536, K=2048)
  k_gemm<false><<<dim3(QLRc / 64, Mrows / 16), blk32, 0, stream>>>(x_bf, wqa_bf, qa_f32, nullptr, QLRc, Dc);
  // ---- rmsnorm(q_a)
  k_rmsnorm<<<Mrows, 256, 0, stream>>>(qa_f32, qnw, qa_bf, QLRc, QLRc);
  // ---- q = q_a_n @ wq_b^T  (4096 x 2048, K=1536)
  k_gemm<false><<<dim3(Hc * DQKc / 64, Mrows / 16), blk32, 0, stream>>>(qa_bf, wqb_bf, q_f32, nullptr, Hc * DQKc, QLRc);
  // ---- kv = x @ wkv_a^T  (4096 x 1088, K=2048)
  k_gemm<false><<<dim3((KVLRc + DRc) / 64, Mrows / 16), blk32, 0, stream>>>(x_bf, wkva_bf, kv_f32, nullptr, KVLRc + DRc, Dc);
  // ---- rmsnorm(c_kv) + rope(k_pe)
  k_rmsnorm<<<Mrows, 256, 0, stream>>>(kv_f32, kvnw, ckv_bf, KVLRc, KVLRc + DRc);
  k_kpe_rope<<<Mrows, 32, 0, stream>>>(kv_f32, fcos, fsin, kpe_bf);
  // ---- kv2 = ckv_n @ wkv_b^T  (4096 x 3072, K=1024), bf16 out
  k_gemm<true><<<dim3(Hc * (DNc + DVc) / 64, Mrows / 16), blk32, 0, stream>>>(ckv_bf, wkvb_bf, nullptr, kv2_bf, Hc * (DNc + DVc), KVLRc);
  // ---- pack Q (rope+scale), K (concat k_pe), V^T
  k_pack_q<<<Mrows * Hc, 64, 0, stream>>>(q_f32, fcos, fsin, Qp);
  k_pack_kv<<<Mrows * Hc, 128, 0, stream>>>(kv2_bf, kpe_bf, Kp, Vt);
  // ---- flash attention
  k_flash<<<dim3(Sc / 16, Bc * Hc), blk32, 0, stream>>>(Qp, Kp, Vt, attn_bf);
  // ---- out = attn @ wo^T  (4096 x 2048, K=2048), f32 out -> d_out
  k_gemm<false><<<dim3(Dc / 64, Mrows / 16), blk32, 0, stream>>>(attn_bf, wo_bf, out, nullptr, Dc, HDc);
}
